// Balancer_53274774340010
// MI455X (gfx1250) — matrix operand order repaired and checked by reading, not compiled, not run on hardware
//
#include <hip/hip_runtime.h>
#include <hip/hip_bf16.h>
#include <stdint.h>

typedef __attribute__((ext_vector_type(16))) _Float16 v16h;
typedef __attribute__((ext_vector_type(8)))  float    v8f;

#define BB 16
#define HH 768
#define WW 2048
#define NBOX 32

#define TILES_X (WW / 16)            // 128
#define TILES_Y (HH / 16)            // 48
#define TILES_PER_IMG (TILES_X * TILES_Y)   // 6144
#define TOTAL_TILES (BB * TILES_PER_IMG)    // 98304
#define WAVES_PER_BLOCK 8
#define MAIN_BLOCKS (TOTAL_TILES / WAVES_PER_BLOCK)  // 12288

// ---------------------------------------------------------------------------
// Kernel 1: per-row / per-column 32-bit box-membership bitmasks.
// rowmask[b*HH+y] bit k = (y >= floor(y1_k)) && (y < ceil(y2_k))
// colmask[b*WW+x] bit k = (x >= floor(x1_k)) && (x < ceil(x2_k))
// ---------------------------------------------------------------------------
__global__ void bal_mask_kernel(const float* __restrict__ boxes,
                                uint32_t* __restrict__ rowmask,
                                uint32_t* __restrict__ colmask) {
    int idx = blockIdx.x * blockDim.x + threadIdx.x;
    const int per_img = HH + WW;
    if (idx >= BB * per_img) return;
    int b = idx / per_img;
    int r = idx - b * per_img;
    const float* bb = boxes + (size_t)b * NBOX * 4;
    uint32_t mask = 0u;
    if (r < HH) {
        int y = r;
        #pragma unroll 4
        for (int i = 0; i < NBOX; ++i) {
            int v1 = (int)floorf(bb[i * 4 + 1]);
            int v2 = (int)ceilf(bb[i * 4 + 3]);
            if (y >= v1 && y < v2) mask |= (1u << i);
        }
        rowmask[b * HH + y] = mask;
    } else {
        int x = r - HH;
        #pragma unroll 4
        for (int i = 0; i < NBOX; ++i) {
            int u1 = (int)floorf(bb[i * 4 + 0]);
            int u2 = (int)ceilf(bb[i * 4 + 2]);
            if (x >= u1 && x < u2) mask |= (1u << i);
        }
        colmask[b * WW + x] = mask;
    }
}

// ---------------------------------------------------------------------------
// Kernel 2: one wave32 per 16x16 pixel tile.
// A(16x32 f16) = row membership, B(32x16 f16) = col membership,
// C = A x B via v_wmma_f32_16x16x32_f16 -> per-pixel covering-box count.
// Fused weighted reduction of the loss tile against the C VGPR layout.
// ---------------------------------------------------------------------------
__global__ void __launch_bounds__(256)
bal_main_kernel(const float* __restrict__ loss,
                const uint32_t* __restrict__ rowmask,
                const uint32_t* __restrict__ colmask,
                float* __restrict__ partials) {
    const int wave = threadIdx.x >> 5;
    const int lane = threadIdx.x & 31;
    const int tile = blockIdx.x * WAVES_PER_BLOCK + wave;

    const int b  = tile / TILES_PER_IMG;
    const int t  = tile - b * TILES_PER_IMG;
    const int ty = t / TILES_X;
    const int tx = t - ty * TILES_X;

    const int hi = lane >> 4;      // half-wave select
    const int m  = lane & 15;      // row (for A) / col (for B,C) within tile

    const uint32_t R  = rowmask[b * HH + ty * 16 + m];
    const uint32_t Cm = colmask[b * WW + tx * 16 + m];

    // Expand bitmasks into WMMA A/B VGPR layouts (ISA 7.12.2, 16-bit, wave32).
    // A 16x32: lanes 0-15 -> V0..3:K0-7, V4..7:K16-23; lanes 16-31 -> K8-15, K24-31
    // B 32x16: same K order as A: lanes 0-15 -> V0..3:K0-7, V4..7:K16-23; lanes 16-31 -> K8-15, K24-31
    union { v16h h; uint32_t u[8]; } Am, Bm;
    #pragma unroll
    for (int j = 0; j < 8; ++j) {
        const int ka = ((j & 4) << 2) + (hi << 3) + ((j & 3) << 1);
        Am.u[j] = (((R >> ka) & 1u) * 0x3C00u) |
                  (((R >> (ka + 1)) & 1u) * 0x3C000000u);
        const int kb = ((j & 4) << 2) + (hi << 3) + ((j & 3) << 1);
        Bm.u[j] = (((Cm >> kb) & 1u) * 0x3C00u) |
                  (((Cm >> (kb + 1)) & 1u) * 0x3C000000u);
    }

    v8f c = {};
    // (neg_a, A, neg_b, B, c_mod, C, reuse_a, reuse_b)
    c = __builtin_amdgcn_wmma_f32_16x16x32_f16(false, Am.h, false, Bm.h,
                                               (short)0, c, false, false);

    // C layout: VGPR r, lanes 0-15 -> (row=r, col=lane); lanes 16-31 -> (row=r+8)
    const float* base = loss + ((size_t)(b * HH + ty * 16 + (hi << 3))) * WW
                             + (size_t)(tx * 16 + m);
    float acc = 0.0f;
    #pragma unroll
    for (int r = 0; r < 8; ++r) {
        const float v = base[(size_t)r * WW];
        acc += v * (c[r] > 0.0f ? 13.0f : 1.0f);
    }

    // Fixed-order wave reduction (deterministic).
    #pragma unroll
    for (int off = 16; off > 0; off >>= 1)
        acc += __shfl_down(acc, off, 32);

    __shared__ float smem[WAVES_PER_BLOCK];
    if (lane == 0) smem[wave] = acc;
    __syncthreads();
    if (threadIdx.x == 0) {
        float s = 0.0f;
        #pragma unroll
        for (int i = 0; i < WAVES_PER_BLOCK; ++i) s += smem[i];
        partials[blockIdx.x] = s;
    }
}

// ---------------------------------------------------------------------------
// Kernel 3: single-block deterministic reduction of block partials.
// ---------------------------------------------------------------------------
__global__ void bal_final_kernel(const float* __restrict__ partials,
                                 int n, float* __restrict__ out) {
    __shared__ float smem[256];
    float acc = 0.0f;
    for (int i = threadIdx.x; i < n; i += 256) acc += partials[i];
    smem[threadIdx.x] = acc;
    __syncthreads();
    for (int off = 128; off > 0; off >>= 1) {
        if (threadIdx.x < (unsigned)off) smem[threadIdx.x] += smem[threadIdx.x + off];
        __syncthreads();
    }
    if (threadIdx.x == 0)
        out[0] = smem[0] / (float)((size_t)BB * HH * WW);
}

extern "C" void kernel_launch(void* const* d_in, const int* in_sizes, int n_in,
                              void* d_out, int out_size, void* d_ws, size_t ws_size,
                              hipStream_t stream) {
    const float* loss  = (const float*)d_in[0];
    const float* boxes = (const float*)d_in[1];
    // d_in[2] = num_gt_per_img (compile-time NBOX == 32 here)

    // Workspace layout
    uint8_t* ws = (uint8_t*)d_ws;
    uint32_t* rowmask = (uint32_t*)ws;                              // BB*HH u32 = 48 KB
    uint32_t* colmask = (uint32_t*)(ws + (size_t)BB * HH * 4);      // BB*WW u32 = 128 KB
    float*    partials = (float*)(ws + (size_t)BB * HH * 4
                                     + (size_t)BB * WW * 4);        // MAIN_BLOCKS f32

    // 1) membership bitmasks
    {
        int total = BB * (HH + WW);
        int blocks = (total + 255) / 256;
        bal_mask_kernel<<<blocks, 256, 0, stream>>>(boxes, rowmask, colmask);
    }
    // 2) WMMA mask + fused weighted streaming reduction
    bal_main_kernel<<<MAIN_BLOCKS, 256, 0, stream>>>(loss, rowmask, colmask, partials);
    // 3) final scalar
    bal_final_kernel<<<1, 256, 0, stream>>>(partials, MAIN_BLOCKS, (float*)d_out);
}
